// SourceAwareAttention_15049565405454
// MI455X (gfx1250) — matrix operand order repaired and checked by reading, not compiled, not run on hardware
//
#include <hip/hip_runtime.h>

// ---------------------------------------------------------------------------
// SourceAwareAttention for MI455X (gfx1250), wave32 + v_wmma_f32_16x16x32_bf16
//   out = softmax(Q K^T / sqrt(D)) @ S  *  V       (B=8, T=2048, D=512, f32 io)
// Round 4: 4-wave workgroups share K/S tiles staged in LDS with
// global_load_async_to_lds_b128 (ASYNCcnt), double-buffered — cuts L2 read
// bandwidth 4x vs per-wave streaming. Buffer selection now via pointer
// arithmetic (pointer arrays of LDS symbols broke the static linker).
// ---------------------------------------------------------------------------

#define BB 8
#define TT 2048
#define DD 512

typedef __attribute__((ext_vector_type(16))) __bf16 bf16x16;
typedef __attribute__((ext_vector_type(8)))  __bf16 bf16x8;
typedef __attribute__((ext_vector_type(8)))  float  f32x8;

__device__ __forceinline__ unsigned short f2bf(float x) {
    unsigned int u = __float_as_uint(x);
    unsigned int r = u + 0x7fffu + ((u >> 16) & 1u);   // round-to-nearest-even
    return (unsigned short)(r >> 16);
}

__device__ __forceinline__ bf16x16 cat8(bf16x8 a, bf16x8 b) {
    return __builtin_shufflevector(a, b, 0,1,2,3,4,5,6,7,8,9,10,11,12,13,14,15);
}

__device__ __forceinline__ bf16x16 ldfrag32(const unsigned short* p) {
    return cat8(*reinterpret_cast<const bf16x8*>(p),
                *reinterpret_cast<const bf16x8*>(p + 16));
}

__device__ __forceinline__ f32x8 wmma_bf16(bf16x16 a, bf16x16 b, f32x8 c) {
    return __builtin_amdgcn_wmma_f32_16x16x32_bf16(false, a, false, b,
                                                   (short)0, c, false, false);
}

// Async copy of one 16-byte chunk, global -> LDS (ASYNCcnt-tracked).
__device__ __forceinline__ void async_cp16(unsigned short* lds_dst,
                                           const unsigned short* gsrc) {
    unsigned lds_off = (unsigned)(uintptr_t)lds_dst;           // LDS byte offset
    unsigned long long ga = (unsigned long long)(uintptr_t)gsrc;
    asm volatile("global_load_async_to_lds_b128 %0, %1, off"
                 :: "v"(lds_off), "v"(ga)
                 : "memory");
}

// ---------------------------------------------------------------------------
// Prep 1: f32 -> bf16 for Q and K (row-major [B][T][D]) — 4 elems / thread
// ---------------------------------------------------------------------------
__global__ __launch_bounds__(256)
void cvt_qk_kernel(const float* __restrict__ q, const float* __restrict__ k,
                   unsigned short* __restrict__ qbf, unsigned short* __restrict__ kbf)
{
    size_t i4 = ((size_t)blockIdx.x * blockDim.x + threadIdx.x) * 4;
    float4 a = *reinterpret_cast<const float4*>(q + i4);
    float4 b = *reinterpret_cast<const float4*>(k + i4);
    ushort4 ra, rb;
    ra.x = f2bf(a.x); ra.y = f2bf(a.y); ra.z = f2bf(a.z); ra.w = f2bf(a.w);
    rb.x = f2bf(b.x); rb.y = f2bf(b.y); rb.z = f2bf(b.z); rb.w = f2bf(b.w);
    *reinterpret_cast<ushort4*>(qbf + i4) = ra;
    *reinterpret_cast<ushort4*>(kbf + i4) = rb;
}

// ---------------------------------------------------------------------------
// Prep 2: S [B][T][D] f32  ->  S^T [B][D][T] bf16 (tiled LDS transpose)
// ---------------------------------------------------------------------------
__global__ __launch_bounds__(256)
void transpose_s_kernel(const float* __restrict__ s, unsigned short* __restrict__ st)
{
    __shared__ unsigned short tile[32][33];
    const int b  = blockIdx.z;
    const int d0 = blockIdx.x * 32;
    const int t0 = blockIdx.y * 32;
    const int tx = threadIdx.x;          // 0..31
    const int ty = threadIdx.y;          // 0..7
    const float* sp = s + (size_t)b * TT * DD;
    unsigned short* op = st + (size_t)b * DD * TT;
#pragma unroll
    for (int i = 0; i < 32; i += 8)
        tile[ty + i][tx] = f2bf(sp[(size_t)(t0 + ty + i) * DD + d0 + tx]);
    __syncthreads();
#pragma unroll
    for (int i = 0; i < 32; i += 8)
        op[(size_t)(d0 + ty + i) * TT + t0 + tx] = tile[tx][ty + i];
}

// ---------------------------------------------------------------------------
// Stage one 32-key tile: K (32x512 bf16, row-major) and S^T (512x32 bf16)
// into LDS with async b128 copies. 128 threads x (16+16) chunks = 64 KB.
// ---------------------------------------------------------------------------
__device__ __forceinline__ void stage_tiles(const unsigned short* __restrict__ kb,
                                            const unsigned short* __restrict__ sb,
                                            unsigned short* kdst, unsigned short* sdst,
                                            int key0, int tid)
{
#pragma unroll
    for (int i = 0; i < 16; ++i) {               // K: 32 rows x 64 chunks
        int q = tid + 128 * i;                   // 0..2047
        int row = q >> 6, cid = q & 63;
        async_cp16(kdst + q * 8, kb + (size_t)(key0 + row) * DD + cid * 8);
    }
#pragma unroll
    for (int i = 0; i < 16; ++i) {               // S^T: 512 rows x 4 chunks
        int q = tid + 128 * i;
        int d = q >> 2, cid = q & 3;
        async_cp16(sdst + q * 8, sb + (size_t)d * TT + key0 + cid * 8);
    }
}

// ---------------------------------------------------------------------------
// Main: 4 waves / workgroup; wave w owns rows [blockIdx.x*64 + w*16, +16).
// All waves share LDS-staged K/S tiles (double-buffered, async-loaded).
//   A-frag (16x32 bf16): lane m=l%16; halves 0-7 -> K = (l/16)*8 + h
//   B-frag (32x16 bf16): lane n=l%16; halves 0-7 -> K = (l/16)*8 + h, 8-15 -> K = 16 + (l/16)*8 + (h-8)
//   C-frag (16x16 f32) : lane l, reg g -> row = g + 8*(l/16), col = l%16
// LDS map (halves): [0,16384) K buf0 | [16384,32768) K buf1 |
//                   [32768,49152) S buf0 | [49152,65536) S buf1 |
//                   [65536,67584) per-wave P scratch (4 x 512)
// ---------------------------------------------------------------------------
__global__ __launch_bounds__(128)
void attn_kernel(const unsigned short* __restrict__ qbf,   // [B][T][D] bf16
                 const unsigned short* __restrict__ kbf,   // [B][T][D] bf16
                 const unsigned short* __restrict__ stbf,  // [B][D][T] bf16
                 const float* __restrict__ v,              // [B][T][D] f32
                 float* __restrict__ out)                  // [B][T][D] f32
{
    extern __shared__ __align__(16) unsigned short smem[];

    const int tid  = threadIdx.x;        // 0..127
    const int w    = tid >> 5;           // wave id 0..3
    const int lane = tid & 31;
    const int half = lane >> 4;
    const int l16  = lane & 15;
    const int b    = blockIdx.y;
    const int row0 = blockIdx.x * 64 + w * 16;
    unsigned short* pl = smem + 65536 + w * 512;   // per-wave P scratch

    const unsigned short* qb = qbf  + (size_t)b * TT * DD;
    const unsigned short* kb = kbf  + (size_t)b * TT * DD;
    const unsigned short* sb = stbf + (size_t)b * DD * TT;

    // ---- Q tile (16 x 512) resident as 16 A-fragments -------------------
    bf16x16 qfrag[16];
#pragma unroll
    for (int c = 0; c < 16; ++c) {
        const unsigned short* base = qb + (size_t)(row0 + l16) * DD + c * 32 + half * 8;
        qfrag[c] = cat8(*reinterpret_cast<const bf16x8*>(base),
                        *reinterpret_cast<const bf16x8*>(base + 16));
    }

    const f32x8 fzero = {0.f, 0.f, 0.f, 0.f, 0.f, 0.f, 0.f, 0.f};
    f32x8 acc[32];
#pragma unroll
    for (int j = 0; j < 32; ++j) acc[j] = fzero;

    float mrow[8], lrow[8];
#pragma unroll
    for (int g = 0; g < 8; ++g) { mrow[g] = -__builtin_inff(); lrow[g] = 0.f; }
    const float scale = 0.044194173824159216f;   // 1/sqrt(512)

    // prologue: stage tile 0 into buffer 0
    stage_tiles(kb, sb, smem, smem + 32768, 0, tid);

    for (int kt = 0; kt < TT / 32; ++kt) {
        const int key0 = kt * 32;

        // current tile ready: my async copies done, then everyone's
        asm volatile("s_wait_asynccnt 0x0" ::: "memory");
        __syncthreads();

        // deep prefetch: issue next tile into the alternate buffer
        if (kt + 1 < TT / 32) {
            const unsigned nb = ((unsigned)(kt + 1) & 1u) * 16384u;
            stage_tiles(kb, sb, smem + nb, smem + 32768 + nb, key0 + 32, tid);
        }

        const unsigned cb = ((unsigned)kt & 1u) * 16384u;
        const unsigned short* klds = smem + cb;
        const unsigned short* slds = smem + 32768 + cb;

        // ---- scores = Q @ K^T (two 16x16 tiles, pipelined LDS frags) ----
        f32x8 sc0 = fzero, sc1 = fzero;
        const unsigned short* kfb = klds + (size_t)l16 * 512 + half * 8;
        bf16x16 k0 = ldfrag32(kfb);
        bf16x16 k1 = ldfrag32(kfb + 16 * 512);
#pragma unroll
        for (int c = 0; c < 16; ++c) {
            bf16x16 n0 = k0, n1 = k1;
            if (c < 15) {
                n0 = ldfrag32(kfb + (c + 1) * 32);
                n1 = ldfrag32(kfb + 16 * 512 + (c + 1) * 32);
            }
            sc0 = wmma_bf16(qfrag[c], k0, sc0);
            sc1 = wmma_bf16(qfrag[c], k1, sc1);
            k0 = n0; k1 = n1;
        }

        // ---- online softmax (per row; rows striped as reg g + 8*half) ---
        float corr[8];
#pragma unroll
        for (int g = 0; g < 8; ++g) {
            float s0 = sc0[g] * scale;
            float s1 = sc1[g] * scale;
            float t = fmaxf(s0, s1);
#pragma unroll
            for (int off = 1; off <= 8; off <<= 1)
                t = fmaxf(t, __shfl_xor(t, off, 32));
            float mnew = fmaxf(mrow[g], t);
            float e0 = __expf(s0 - mnew);
            float e1 = __expf(s1 - mnew);
            sc0[g] = e0;
            sc1[g] = e1;
            float rs = e0 + e1;
#pragma unroll
            for (int off = 1; off <= 8; off <<= 1)
                rs += __shfl_xor(rs, off, 32);
            float cg = __expf(mrow[g] - mnew);
            lrow[g] = lrow[g] * cg + rs;
            mrow[g] = mnew;
            corr[g] = cg;
        }

        // ---- P : C-layout -> LDS first (DS latency hides under rescale) -
#pragma unroll
        for (int g = 0; g < 8; ++g) {
            int r = g + half * 8;
            pl[r * 32 + l16]      = f2bf(sc0[g]);
            pl[r * 32 + 16 + l16] = f2bf(sc1[g]);
        }

        // ---- rescale running output accumulator (overlaps DS stores) ---
#pragma unroll
        for (int j = 0; j < 32; ++j)
#pragma unroll
            for (int g = 0; g < 8; ++g) acc[j][g] *= corr[g];

        asm volatile("s_wait_dscnt 0" ::: "memory");
        const unsigned short* pbp = pl + l16 * 32 + half * 8;
        bf16x16 pfrag = cat8(*reinterpret_cast<const bf16x8*>(pbp),
                             *reinterpret_cast<const bf16x8*>(pbp + 16));
        asm volatile("" ::: "memory");

        // ---- O += P @ S (S^T layout in LDS; pipelined frags) ------------
        const unsigned short* sfb = slds + (size_t)l16 * 32 + half * 8;
        bf16x16 sB = ldfrag32(sfb);
#pragma unroll
        for (int j = 0; j < 32; ++j) {
            bf16x16 sN = sB;
            if (j < 31)
                sN = ldfrag32(sfb + (j + 1) * 16 * 32);
            acc[j] = wmma_bf16(pfrag, sB, acc[j]);
            sB = sN;
        }
    }

    // ---- epilogue: normalize, modulate by V, store f32 ------------------
    float invl[8];
#pragma unroll
    for (int g = 0; g < 8; ++g) invl[g] = 1.0f / lrow[g];

    const float* vb = v   + (size_t)b * TT * DD;
    float*       ob = out + (size_t)b * TT * DD;
#pragma unroll
    for (int j = 0; j < 32; ++j) {
        int dcol = j * 16 + l16;
#pragma unroll
        for (int g = 0; g < 8; ++g) {
            size_t idx = (size_t)(row0 + g + half * 8) * DD + dcol;
            ob[idx] = acc[j][g] * invl[g] * vb[idx];
        }
    }
}

// ---------------------------------------------------------------------------
extern "C" void kernel_launch(void* const* d_in, const int* in_sizes, int n_in,
                              void* d_out, int out_size, void* d_ws, size_t ws_size,
                              hipStream_t stream)
{
    const float* q = (const float*)d_in[0];
    const float* k = (const float*)d_in[1];
    const float* v = (const float*)d_in[2];
    const float* s = (const float*)d_in[3];
    float* out = (float*)d_out;

    const size_t n = (size_t)BB * TT * DD;          // 8M elements
    unsigned short* qbf  = (unsigned short*)d_ws;   // bf16 Q  [B][T][D]
    unsigned short* kbf  = qbf + n;                 // bf16 K  [B][T][D]
    unsigned short* stbf = kbf + n;                 // bf16 S^T [B][D][T]

    // Prep: convert + transpose (streams ~150 MB, negligible vs 69 GFLOP)
    cvt_qk_kernel<<<dim3((unsigned)(n / 4 / 256)), dim3(256), 0, stream>>>(q, k, qbf, kbf);
    transpose_s_kernel<<<dim3(DD / 32, TT / 32, BB), dim3(32, 8, 1), 0, stream>>>(s, stbf);

    // Main: 4 waves/WG, 64 rows per WG, 132 KB dynamic LDS (double buffers)
    const size_t lds_bytes = (2 * 16384 + 2 * 16384 + 4 * 512) * sizeof(unsigned short);
    attn_kernel<<<dim3(TT / 64, BB, 1), dim3(128, 1, 1), lds_bytes, stream>>>(
        qbf, kbf, stbf, v, out);
}
